// CnvEncoder_28965259444914
// MI455X (gfx1250) — compile-verified, matrix operand order checked by reading, not run on hardware
//
#include <hip/hip_runtime.h>
#include <hip/hip_bf16.h>

typedef __attribute__((ext_vector_type(16))) _Float16 v16h;
typedef __attribute__((ext_vector_type(8)))  _Float16 v8h;
typedef __attribute__((ext_vector_type(4)))  _Float16 v4h;
typedef __attribute__((ext_vector_type(8)))  float    v8f;

#define D_MODEL 512
#define LATENT  768
#define D_STATE 16
#define D_CONV  4
#define D_INNER 1536
#define DT_RANK 48
#define BATCH   8
#define SEQ     2048
#define MTOT    (BATCH*SEQ)        // 16384 rows for all token-parallel GEMMs

// ------------------------------------------------------------------
// WMMA GEMM:  C[M x N] = A[M x K] (row-major, stride lda)
//                      * W[N x K]^T (row-major, stride ldw)  (+bias)(+softplus)
// Block tile 128x128, 8 waves arranged 4(M) x 2(N). Each wave: 2 row tiles
// x 4 col tiles = 8 v_wmma_f32_16x16x32_f16 per 32-wide K slab, from
// 2 A-fragments + 4 B-fragments (each B reused by both row tiles):
// 1.5 ds_load_b128 per WMMA. Double-buffered LDS, software-pipelined:
// global loads for slab k+1 issue right after the single barrier and are
// consumed only at the next iteration's LDS-store phase.
// GUARD=true: branchless clamped loads + select (for N=80 / K=48 GEMMs).
// ------------------------------------------------------------------
#define BM 128
#define BN 128
#define BK 32

template<bool GUARD>
__global__ __launch_bounds__(256)
void wmma_gemm_nt(const float* __restrict__ A, int lda,
                  const float* __restrict__ W, int ldw,
                  const float* __restrict__ bias,
                  float* __restrict__ C, int ldc,
                  int N, int K, int act)
{
  __shared__ _Float16 As[2][BM][BK + 8];   // stride 40 halves = 80B (16B aligned)
  __shared__ _Float16 Ws[2][BN][BK + 8];

  const int tid  = threadIdx.x;
  const int lane = tid & 31;
  const int wv   = tid >> 5;                 // 0..7
  const int wrow = wv & 3;                   // 4 wave-rows  (32 rows each)
  const int wcol = wv >> 2;                  // 2 wave-cols  (64 cols each)
  const int rowBase = blockIdx.y * BM;
  const int colBase = blockIdx.x * BN;

  const int lr   = lane & 15;                // row (A) / col (B) within tile
  const int koff = (lane >> 4) << 3;         // 0 or 8 (ISA f16 A/B lane layout)

  v8f acc[2][4] = {};
  float4 areg[4];                            // staged A slab (4 float4 / thread)
  float4 wreg[4];                            // staged W slab (4 float4 / thread)

  // Per-thread staging coordinates: slot j covers LDS element i = tid + 256*j.
  // r = i/8 (row), c4 = (i%8)*4 (K offset within 32-wide slab).
  auto load_tiles = [&](int kk) {
    #pragma unroll
    for (int j = 0; j < 4; ++j) {
      int i = tid + 256 * j, r = i >> 3, c4 = (i & 7) << 2;
      if (GUARD) {
        int k  = kk + c4;
        int kc = (k < K - 4) ? k : (K - 4);          // clamped, always in-bounds
        float4 t = *(const float4*)&A[(size_t)(rowBase + r) * lda + kc];
        areg[j] = (k < K) ? t : make_float4(0.f, 0.f, 0.f, 0.f);
      } else {
        areg[j] = *(const float4*)&A[(size_t)(rowBase + r) * lda + kk + c4];
      }
    }
    #pragma unroll
    for (int j = 0; j < 4; ++j) {
      int i = tid + 256 * j, r = i >> 3, c4 = (i & 7) << 2;
      if (GUARD) {
        int n = colBase + r, k = kk + c4;
        int nc = (n < N) ? n : (N - 1);
        int kc = (k < K - 4) ? k : (K - 4);
        float4 t = *(const float4*)&W[(size_t)nc * ldw + kc];
        wreg[j] = (n < N && k < K) ? t : make_float4(0.f, 0.f, 0.f, 0.f);
      } else {
        wreg[j] = *(const float4*)&W[(size_t)(colBase + r) * ldw + kk + c4];
      }
    }
  };
  auto store_tiles = [&](int p) {
    #pragma unroll
    for (int j = 0; j < 4; ++j) {
      int i = tid + 256 * j, r = i >> 3, c4 = (i & 7) << 2;
      float4 v = areg[j];
      v4h h4 = { (_Float16)v.x, (_Float16)v.y, (_Float16)v.z, (_Float16)v.w };
      *(v4h*)&As[p][r][c4] = h4;
    }
    #pragma unroll
    for (int j = 0; j < 4; ++j) {
      int i = tid + 256 * j, r = i >> 3, c4 = (i & 7) << 2;
      float4 v = wreg[j];
      v4h h4 = { (_Float16)v.x, (_Float16)v.y, (_Float16)v.z, (_Float16)v.w };
      *(v4h*)&Ws[p][r][c4] = h4;
    }
  };
  // Fragment loader: lane<16 holds K {0..7,16..23}, lane>=16 {8..15,24..31}
  auto load_frag = [&](const _Float16* base) -> v16h {
    v8h lo = *(const v8h*)(base + koff);
    v8h hi = *(const v8h*)(base + 16 + koff);
    v16h f;
    #pragma unroll
    for (int i = 0; i < 8; ++i) { f[i] = lo[i]; f[8 + i] = hi[i]; }
    return f;
  };

  load_tiles(0);
  int p = 0;
  for (int kk = 0; kk < K; kk += BK, p ^= 1) {
    store_tiles(p);
    __syncthreads();
    if (kk + BK < K) load_tiles(kk + BK);    // prefetch next slab into regs

    v16h a0 = load_frag(&As[p][(wrow << 5) + lr][0]);
    v16h a1 = load_frag(&As[p][(wrow << 5) + 16 + lr][0]);
    #pragma unroll
    for (int ct = 0; ct < 4; ++ct) {
      v16h bf = load_frag(&Ws[p][(wcol << 6) + (ct << 4) + lr][0]);
      acc[0][ct] = __builtin_amdgcn_wmma_f32_16x16x32_f16(
          false, a0, false, bf, (short)0, acc[0][ct], false, false);
      acc[1][ct] = __builtin_amdgcn_wmma_f32_16x16x32_f16(
          false, a1, false, bf, (short)0, acc[1][ct], false, false);
    }
  }

  // ---- epilogue: C/D layout: VGPR v -> row v + 8*(lane>=16), col lane%16 ----
  #pragma unroll
  for (int rt = 0; rt < 2; ++rt) {
    #pragma unroll
    for (int ct = 0; ct < 4; ++ct) {
      int col = colBase + (wcol << 6) + (ct << 4) + lr;
      if (!GUARD || col < N) {
        float bv = bias ? bias[col] : 0.0f;
        #pragma unroll
        for (int v = 0; v < 8; ++v) {
          int row = rowBase + (wrow << 5) + (rt << 4) + v + ((lane >> 4) << 3);
          float val = acc[rt][ct][v] + bv;
          if (act == 1) val = (val > 20.0f) ? val : log1pf(__expf(val));  // softplus
          C[(size_t)row * ldc + col] = val;
        }
      }
    }
  }
}

// ------------------------------------------------------------------
// Depthwise causal conv1d (k=4) + SiLU.  u = first half of xz (stride 3072).
// ------------------------------------------------------------------
__global__ __launch_bounds__(256)
void conv_silu(const float* __restrict__ u,     // (B*L, 3072), cols [0,1536)
               const float* __restrict__ cw,    // (1536, 4)
               const float* __restrict__ cb,    // (1536)
               float* __restrict__ uc)          // (B*L, 1536)
{
  long long idx = (long long)blockIdx.x * 256 + threadIdx.x;
  if (idx >= (long long)MTOT * D_INNER) return;
  int d = (int)(idx % D_INNER);
  long long bl = idx / D_INNER;                 // b*SEQ + l
  int l = (int)(bl % SEQ);
  long long bbase = bl - l;                     // b*SEQ
  float acc = cb[d];
  #pragma unroll
  for (int k = 0; k < D_CONV; ++k) {
    int li = l - (D_CONV - 1) + k;
    if (li >= 0) acc += cw[d * D_CONV + k] * u[(bbase + li) * 3072 + d];
  }
  acc = acc / (1.0f + __expf(-acc));            // SiLU
  uc[idx] = acc;
}

// ------------------------------------------------------------------
// Selective scan: one thread per (batch, channel). 16-float state in regs,
// A = -exp(A_log) hoisted. B/C (32 floats/step, shared per batch) via float4
// loads served by cache broadcast. Gating by silu(z) fused. Writes yg
// (aliases dt: each element is read then overwritten by the same thread).
// ------------------------------------------------------------------
__global__ __launch_bounds__(256)
void ssm_scan(const float* dt,                   // (B*L, 1536)
              const float* __restrict__ uc,      // (B*L, 1536)
              const float* __restrict__ xz,      // (B*L, 3072); z = cols [1536,3072)
              const float* __restrict__ dbc,     // (B*L, 80); B@48, C@64
              const float* __restrict__ A_log,   // (1536, 16)
              const float* __restrict__ Dv,      // (1536)
              float* yg)                          // (B*L, 1536), aliases dt
{
  int d = blockIdx.x * 256 + threadIdx.x;        // channel
  int b = blockIdx.y;                            // batch
  float Aa[D_STATE];
  #pragma unroll
  for (int n = 0; n < D_STATE; ++n) Aa[n] = -__expf(A_log[d * D_STATE + n]);
  float Dd = Dv[d];
  float st[D_STATE];
  #pragma unroll
  for (int n = 0; n < D_STATE; ++n) st[n] = 0.0f;

  const long long rbase = (long long)b * SEQ;
  for (int l = 0; l < SEQ; ++l) {
    long long r = rbase + l;
    float dtv = dt[r * D_INNER + d];
    float uv  = uc[r * D_INNER + d];
    float zv  = xz[r * 3072 + 1536 + d];
    float Bv[D_STATE], Cv[D_STATE];
    #pragma unroll
    for (int n = 0; n < D_STATE; n += 4) {
      float4 tb = *(const float4*)(dbc + r * 80 + 48 + n);
      Bv[n] = tb.x; Bv[n + 1] = tb.y; Bv[n + 2] = tb.z; Bv[n + 3] = tb.w;
      float4 tc = *(const float4*)(dbc + r * 80 + 64 + n);
      Cv[n] = tc.x; Cv[n + 1] = tc.y; Cv[n + 2] = tc.z; Cv[n + 3] = tc.w;
    }
    float du = dtv * uv;
    float y = 0.0f;
    #pragma unroll
    for (int n = 0; n < D_STATE; ++n) {
      float dA = __expf(dtv * Aa[n]);
      st[n] = dA * st[n] + du * Bv[n];
      y += st[n] * Cv[n];
    }
    y += Dd * uv;
    y *= zv / (1.0f + __expf(-zv));              // * silu(z)
    yg[r * D_INNER + d] = y;
  }
}

// ------------------------------------------------------------------
// ybar[b,d] = mean_l yg[b,l,d]   (mean commutes with out_proj)
// ------------------------------------------------------------------
__global__ __launch_bounds__(256)
void col_mean(const float* __restrict__ yg, float* __restrict__ ybar)
{
  int d = blockIdx.x * 256 + threadIdx.x;
  int b = blockIdx.y;
  const float* p = yg + (long long)b * SEQ * D_INNER + d;
  float s = 0.0f;
  for (int l = 0; l < SEQ; ++l) s += p[(long long)l * D_INNER];
  ybar[b * D_INNER + d] = s * (1.0f / SEQ);
}

// ------------------------------------------------------------------
// out[b,n] = ybar[b,:] . w_out[n,:]   (tiny: 8x768x1536)
// ------------------------------------------------------------------
__global__ __launch_bounds__(256)
void out_proj(const float* __restrict__ ybar, const float* __restrict__ wout,
              float* __restrict__ out)
{
  int idx = blockIdx.x * 256 + threadIdx.x;      // b*768 + n
  if (idx >= BATCH * LATENT) return;
  int b = idx / LATENT, n = idx % LATENT;
  const float* y = ybar + b * D_INNER;
  const float* w = wout + (size_t)n * D_INNER;
  float s = 0.0f;
  for (int k = 0; k < D_INNER; k += 4) {
    float4 a  = *(const float4*)(y + k);
    float4 ww = *(const float4*)(w + k);
    s += a.x * ww.x + a.y * ww.y + a.z * ww.z + a.w * ww.w;
  }
  out[idx] = s;
}

// ------------------------------------------------------------------
extern "C" void kernel_launch(void* const* d_in, const int* in_sizes, int n_in,
                              void* d_out, int out_size, void* d_ws, size_t ws_size,
                              hipStream_t stream) {
  const float* x      = (const float*)d_in[0];
  const float* w_ip   = (const float*)d_in[1];
  const float* b_ip   = (const float*)d_in[2];
  const float* w_in   = (const float*)d_in[3];
  const float* conv_w = (const float*)d_in[4];
  const float* conv_b = (const float*)d_in[5];
  const float* w_xp   = (const float*)d_in[6];
  const float* w_dt   = (const float*)d_in[7];
  const float* b_dt   = (const float*)d_in[8];
  const float* A_log  = (const float*)d_in[9];
  const float* Dv     = (const float*)d_in[10];
  const float* w_out  = (const float*)d_in[11];
  float* out = (float*)d_out;

  // Workspace layout (bytes). 'h' is dead after GEMM2; 'uc' reuses its region.
  char* ws = (char*)d_ws;
  float* xz   = (float*)(ws);                          // 16384*3072*4 = 201326592
  float* h    = (float*)(ws + 201326592ULL);           // 16384* 768*4 =  50331648
  float* uc   = (float*)(ws + 201326592ULL);           // reuse: 16384*1536*4 = 100663296
  float* dbc  = (float*)(ws + 301989888ULL);           // 16384*  80*4 =   5242880
  float* dt   = (float*)(ws + 307232768ULL);           // 16384*1536*4 = 100663296 (also yg)
  float* ybar = (float*)(ws + 407896064ULL);           // 8*1536*4
  (void)in_sizes; (void)n_in; (void)out_size; (void)ws_size;

  dim3 blk(256);

  // 1) h = x @ w_ip^T + b_ip      (M=16384, N=768, K=512)   [fast path]
  wmma_gemm_nt<false><<<dim3(LATENT / BN, MTOT / BM), blk, 0, stream>>>(
      x, D_MODEL, w_ip, D_MODEL, b_ip, h, LATENT, LATENT, D_MODEL, 0);

  // 2) xz = h @ w_in^T            (M=16384, N=3072, K=768)  [fast path]
  wmma_gemm_nt<false><<<dim3(3072 / BN, MTOT / BM), blk, 0, stream>>>(
      h, LATENT, w_in, LATENT, nullptr, xz, 3072, 3072, LATENT, 0);

  // 3) uc = silu(causal depthwise conv(u))  ; u = xz[:, :1536]
  conv_silu<<<(int)(((long long)MTOT * D_INNER + 255) / 256), blk, 0, stream>>>(
      xz, conv_w, conv_b, uc);

  // 4) dbc = uc @ w_xp^T          (M=16384, N=80, K=1536)   [guarded: N]
  wmma_gemm_nt<true><<<dim3(1, MTOT / BM), blk, 0, stream>>>(
      uc, D_INNER, w_xp, D_INNER, nullptr, dbc, 80, 80, D_INNER, 0);

  // 5) dt = softplus(dbc[:, :48] @ w_dt^T + b_dt)  (M=16384, N=1536, K=48) [guarded: K]
  wmma_gemm_nt<true><<<dim3(D_INNER / BN, MTOT / BM), blk, 0, stream>>>(
      dbc, 80, w_dt, DT_RANK, b_dt, dt, D_INNER, D_INNER, DT_RANK, 1);

  // 6) selective scan + skip (D*uc) + silu(z) gating ; writes yg in-place over dt
  ssm_scan<<<dim3(D_INNER / 256, BATCH), blk, 0, stream>>>(
      dt, uc, xz, dbc, A_log, Dv, dt);

  // 7) ybar = mean_L(yg);  out = ybar @ w_out^T
  col_mean<<<dim3(D_INNER / 256, BATCH), blk, 0, stream>>>(dt, ybar);
  out_proj<<<(BATCH * LATENT + 255) / 256, blk, 0, stream>>>(ybar, w_out, out);
}
